// SqnxtUnit_74741020885311
// MI455X (gfx1250) — 8-bit matrix operand order repaired and checked by reading, compiled, not run on hardware
//
#include <hip/hip_runtime.h>
#include <stdint.h>

typedef __attribute__((ext_vector_type(8))) int v8i;

#define MTOT 50176
#define HH 28
#define WW 28
#define HW 784

#if defined(__gfx1250__) && __has_builtin(__builtin_amdgcn_global_load_async_to_lds_b128)
#define USE_ASYNC_LDS 1
typedef int v4i __attribute__((vector_size(16)));
typedef __attribute__((address_space(1))) v4i gv4i;
typedef __attribute__((address_space(3))) v4i lv4i;
__device__ __forceinline__ void wait_async0() {
#if __has_builtin(__builtin_amdgcn_s_wait_asynccnt)
  __builtin_amdgcn_s_wait_asynccnt(0);
#else
  asm volatile("s_wait_asynccnt 0x0" ::: "memory");
#endif
}
#else
#define USE_ASYNC_LDS 0
#endif

__device__ __forceinline__ float wave_max(float v) {
#pragma unroll
  for (int off = 16; off >= 1; off >>= 1)
    v = fmaxf(v, __shfl_xor(v, off, 32));
  return v;
}

__device__ __forceinline__ signed char fq(float v, float amax) {
  float s = fmaxf(amax, 1e-8f) * (1.0f / 7.0f);
  float q = rintf(v / s);
  q = fminf(7.0f, fmaxf(-7.0f, q));
  return (signed char)q;
}

__global__ void k_zero_amax(unsigned* a) {
  if (threadIdx.x < 16) a[threadIdx.x] = 0u;
}

__global__ void k_amax(const float* __restrict__ t, int n, unsigned* __restrict__ out) {
  float m = 0.0f;
  for (int i = blockIdx.x * blockDim.x + threadIdx.x; i < n; i += gridDim.x * blockDim.x)
    m = fmaxf(m, fabsf(t[i]));
  m = wave_max(m);
  __shared__ float red[8];
  int lane = threadIdx.x & 31, wid = threadIdx.x >> 5;
  if (lane == 0) red[wid] = m;
  __syncthreads();
  if (wid == 0) {
    m = (lane < (int)(blockDim.x >> 5)) ? red[lane] : 0.0f;
    m = wave_max(m);
    if (lane == 0) atomicMax(out, __float_as_uint(m));
  }
}

__global__ void k_quant_w(const float* __restrict__ w, signed char* __restrict__ out,
                          int Co, int Ci, int KH, int KW, const unsigned* __restrict__ amax) {
  int K = Ci * KH * KW;
  int t = blockIdx.x * blockDim.x + threadIdx.x;
  if (t >= Co * K) return;
  float a = __uint_as_float(*amax);
  int co = t / K, kidx = t % K;
  int c = kidx % Ci, tmp = kidx / Ci;
  int kw = tmp % KW, kh = tmp / KW;
  out[t] = fq(w[((co * Ci + c) * KH + kh) * KW + kw], a);
}

__global__ void k_quant_x(const float* __restrict__ x, signed char* __restrict__ out,
                          const unsigned* __restrict__ amax) {
  int t = blockIdx.x * blockDim.x + threadIdx.x;
  if (t >= MTOT * 512) return;
  float a = __uint_as_float(*amax);
  int c = t & 511;
  int m = t >> 9;
  out[t] = fq(x[((size_t)((m / HW) * 512 + c)) * HW + (m % HW)], a);
}

__global__ void k_im2col_q(const float* __restrict__ in, signed char* __restrict__ out,
                           int C, int KH, int KW, int padH, int padW, int total,
                           const unsigned* __restrict__ amax) {
  int t = blockIdx.x * blockDim.x + threadIdx.x;
  if (t >= total) return;
  int K = C * KH * KW;
  float a = __uint_as_float(*amax);
  int m = t / K, kidx = t % K;
  int c = kidx % C, tmp = kidx / C;
  int kw = tmp % KW, kh = tmp / KW;
  int b = m / HW, rem = m % HW;
  int h = rem / WW, w = rem % WW;
  int ih = h + kh - padH, iw = w + kw - padW;
  signed char q = 0;
  if (ih >= 0 && ih < HH && iw >= 0 && iw < WW)
    q = fq(in[((size_t)((b * HH + ih) * WW + iw)) * C + c], a);
  out[t] = q;
}

#define LDA 80
__global__ __launch_bounds__(256) void k_gemm_i8(
    const signed char* __restrict__ A, const signed char* __restrict__ Bw,
    const float* __restrict__ bias,
    const unsigned* __restrict__ amaxA, const unsigned* __restrict__ amaxW,
    float* __restrict__ out_nhwc, unsigned* __restrict__ amax_out,
    const float* __restrict__ identity_nchw, float* __restrict__ out_final,
    int K, int N, int final_mode) {
  __shared__ __align__(16) signed char ldsA[2][128 * LDA];
  __shared__ __align__(16) signed char ldsB[2][128 * LDA];

  const int tid = threadIdx.x;
  const int lane = tid & 31, wid = tid >> 5;
  const int lh = lane >> 4, lm = lane & 15;
  const int waveM = wid & 1, waveN = wid >> 1;
  const int Mb = blockIdx.x * 128, Nb = blockIdx.y * 128;

  v8i acc[4][2];
#pragma unroll
  for (int i = 0; i < 4; ++i)
#pragma unroll
    for (int j = 0; j < 2; ++j)
#pragma unroll
      for (int r = 0; r < 8; ++r) acc[i][j][r] = 0;

  auto issue_tile = [&](int kk, int buf) {
    signed char* dA = &ldsA[buf][0];
    signed char* dB = &ldsB[buf][0];
#pragma unroll
    for (int i = 0; i < 2; ++i) {
      int c = tid + i * 256;
      int r = c >> 2, off = (c & 3) << 4;
      const signed char* gA = A + (size_t)(Mb + r) * K + kk + off;
      const signed char* gB = Bw + (size_t)(Nb + r) * K + kk + off;
#if USE_ASYNC_LDS
      __builtin_amdgcn_global_load_async_to_lds_b128((gv4i*)gA, (lv4i*)(dA + r * LDA + off), 0, 0);
      __builtin_amdgcn_global_load_async_to_lds_b128((gv4i*)gB, (lv4i*)(dB + r * LDA + off), 0, 0);
#else
      *(int4*)(dA + r * LDA + off) = *(const int4*)gA;
      *(int4*)(dB + r * LDA + off) = *(const int4*)gB;
#endif
    }
  };

  issue_tile(0, 0);
  int cur = 0;
#if USE_ASYNC_LDS
  wait_async0();
#endif
  __syncthreads();

  for (int kk = 0; kk < K; kk += 64) {
    const int nxt = cur ^ 1;
    if (kk + 64 < K) issue_tile(kk + 64, nxt);

    const signed char* curA = &ldsA[cur][0];
    const signed char* curB = &ldsB[cur][0];
    v8i af[4], bf[2];
#pragma unroll
    for (int s = 0; s < 4; ++s) {
      const signed char* abase = curA + (waveM * 64 + s * 16 + lm) * LDA;
#pragma unroll
      for (int j = 0; j < 8; ++j)
        af[s][j] = *(const int*)(abase + ((j >> 2) << 5) + (lh << 4) + ((j & 3) << 2));
    }
#pragma unroll
    for (int s = 0; s < 2; ++s) {
      const signed char* bbase = curB + (waveN * 32 + s * 16 + lm) * LDA;
#pragma unroll
      for (int j = 0; j < 8; ++j)
        bf[s][j] = *(const int*)(bbase + ((j >> 2) << 5) + (lh << 4) + ((j & 3) << 2));
    }

#pragma unroll
    for (int i = 0; i < 4; ++i)
#pragma unroll
      for (int j = 0; j < 2; ++j)
        acc[i][j] = __builtin_amdgcn_wmma_i32_16x16x64_iu8(
            true, af[i], true, bf[j], acc[i][j], false, false);

#if USE_ASYNC_LDS
    wait_async0();
#endif
    __syncthreads();
    cur = nxt;
  }

  const float sa = fmaxf(__uint_as_float(*amaxA), 1e-8f) * (1.0f / 7.0f);
  const float sw = fmaxf(__uint_as_float(*amaxW), 1e-8f) * (1.0f / 7.0f);
  const float scale = sa * sw;
  float lmax = 0.0f;
#pragma unroll
  for (int i = 0; i < 4; ++i) {
#pragma unroll
    for (int j = 0; j < 2; ++j) {
      const int n = Nb + waveN * 32 + j * 16 + lm;
      const float bv = bias[n];
#pragma unroll
      for (int r = 0; r < 8; ++r) {
        const int m = Mb + waveM * 64 + i * 16 + lh * 8 + r;
        float v = (float)acc[i][j][r] * scale + bv;
        if (final_mode) {
          const int b = m / HW, rem = m % HW;
          const size_t idx = ((size_t)(b * 512 + n)) * HW + rem;
          out_final[idx] = fmaxf(v + identity_nchw[idx], 0.0f);
        } else {
          out_nhwc[(size_t)m * N + n] = v;
          lmax = fmaxf(lmax, fabsf(v));
        }
      }
    }
  }
  if (!final_mode) {
    lmax = wave_max(lmax);
    if (lane == 0) atomicMax(amax_out, __float_as_uint(lmax));
  }
}

extern "C" void kernel_launch(void* const* d_in, const int* in_sizes, int n_in,
                              void* d_out, int out_size, void* d_ws, size_t ws_size,
                              hipStream_t stream) {
  (void)in_sizes; (void)n_in; (void)out_size; (void)ws_size;
  const float* x  = (const float*)d_in[0];
  const float* w1 = (const float*)d_in[1];
  const float* b1 = (const float*)d_in[2];
  const float* w2 = (const float*)d_in[3];
  const float* b2 = (const float*)d_in[4];
  const float* w3 = (const float*)d_in[5];
  const float* b3 = (const float*)d_in[6];
  const float* w4 = (const float*)d_in[7];
  const float* b4 = (const float*)d_in[8];
  const float* w5 = (const float*)d_in[9];
  const float* b5 = (const float*)d_in[10];
  float* out = (float*)d_out;

  char* ws = (char*)d_ws;
  size_t o = 0;
  auto alloc = [&](size_t sz) { size_t r = o; o += (sz + 255) & ~(size_t)255; return r; };
  const size_t o_amax = alloc(64);
  const size_t o_wq1 = alloc(256 * 512);
  const size_t o_wq2 = alloc(128 * 256);
  const size_t o_wq3 = alloc(256 * 384);
  const size_t o_wq4 = alloc(256 * 768);
  const size_t o_wq5 = alloc(512 * 256);
  const size_t o_Aa  = alloc((size_t)MTOT * 512);
  const size_t o_Ab  = alloc((size_t)MTOT * 768);
  const size_t o_F1  = alloc((size_t)MTOT * 256 * 4);
  const size_t o_F2  = alloc((size_t)MTOT * 256 * 4);

  unsigned* amax = (unsigned*)(ws + o_amax);
  signed char* wq1 = (signed char*)(ws + o_wq1);
  signed char* wq2 = (signed char*)(ws + o_wq2);
  signed char* wq3 = (signed char*)(ws + o_wq3);
  signed char* wq4 = (signed char*)(ws + o_wq4);
  signed char* wq5 = (signed char*)(ws + o_wq5);
  signed char* Aa  = (signed char*)(ws + o_Aa);
  signed char* Ab  = (signed char*)(ws + o_Ab);
  float* F1 = (float*)(ws + o_F1);
  float* F2 = (float*)(ws + o_F2);

  k_zero_amax<<<1, 64, 0, stream>>>(amax);
  k_amax<<<2048, 256, 0, stream>>>(x,  MTOT * 512,   amax + 0);
  k_amax<<<64,   256, 0, stream>>>(w1, 256 * 512,    amax + 1);
  k_amax<<<16,   256, 0, stream>>>(w2, 128 * 256,    amax + 2);
  k_amax<<<32,   256, 0, stream>>>(w3, 256 * 128 * 3, amax + 3);
  k_amax<<<64,   256, 0, stream>>>(w4, 256 * 256 * 3, amax + 4);
  k_amax<<<64,   256, 0, stream>>>(w5, 512 * 256,    amax + 5);

  k_quant_w<<<(256 * 512 + 255) / 256, 256, 0, stream>>>(w1, wq1, 256, 512, 1, 1, amax + 1);
  k_quant_w<<<(128 * 256 + 255) / 256, 256, 0, stream>>>(w2, wq2, 128, 256, 1, 1, amax + 2);
  k_quant_w<<<(256 * 384 + 255) / 256, 256, 0, stream>>>(w3, wq3, 256, 128, 1, 3, amax + 3);
  k_quant_w<<<(256 * 768 + 255) / 256, 256, 0, stream>>>(w4, wq4, 256, 256, 3, 1, amax + 4);
  k_quant_w<<<(512 * 256 + 255) / 256, 256, 0, stream>>>(w5, wq5, 512, 256, 1, 1, amax + 5);

  k_quant_x<<<(MTOT * 512) / 256, 256, 0, stream>>>(x, Aa, amax + 0);
  { dim3 g(MTOT / 128, 256 / 128);
    k_gemm_i8<<<g, 256, 0, stream>>>(Aa, wq1, b1, amax + 0, amax + 1,
                                     F1, amax + 6, nullptr, nullptr, 512, 256, 0); }
  k_im2col_q<<<(MTOT * 256) / 256, 256, 0, stream>>>(F1, Ab, 256, 1, 1, 0, 0, MTOT * 256, amax + 6);
  { dim3 g(MTOT / 128, 128 / 128);
    k_gemm_i8<<<g, 256, 0, stream>>>(Ab, wq2, b2, amax + 6, amax + 2,
                                     F2, amax + 7, nullptr, nullptr, 256, 128, 0); }
  k_im2col_q<<<(MTOT * 384) / 256, 256, 0, stream>>>(F2, Aa, 128, 1, 3, 0, 1, MTOT * 384, amax + 7);
  { dim3 g(MTOT / 128, 256 / 128);
    k_gemm_i8<<<g, 256, 0, stream>>>(Aa, wq3, b3, amax + 7, amax + 3,
                                     F1, amax + 8, nullptr, nullptr, 384, 256, 0); }
  k_im2col_q<<<(MTOT * 768) / 256, 256, 0, stream>>>(F1, Ab, 256, 3, 1, 1, 0, MTOT * 768, amax + 8);
  { dim3 g(MTOT / 128, 256 / 128);
    k_gemm_i8<<<g, 256, 0, stream>>>(Ab, wq4, b4, amax + 8, amax + 4,
                                     F2, amax + 9, nullptr, nullptr, 768, 256, 0); }
  k_im2col_q<<<(MTOT * 256) / 256, 256, 0, stream>>>(F2, Aa, 256, 1, 1, 0, 0, MTOT * 256, amax + 9);
  { dim3 g(MTOT / 128, 512 / 128);
    k_gemm_i8<<<g, 256, 0, stream>>>(Aa, wq5, b5, amax + 9, amax + 5,
                                     nullptr, nullptr, x, out, 256, 512, 1); }
}
